// WindBiasGNN_55886114456025
// MI455X (gfx1250) — matrix operand order repaired and checked by reading, not compiled, not run on hardware
//
#include <hip/hip_runtime.h>
#include <hip/hip_bf16.h>

// ---------------- WMMA helpers (CDNA5 / gfx1250, wave32) ----------------
typedef __attribute__((ext_vector_type(16))) _Float16 v16h;
typedef __attribute__((ext_vector_type(8)))  float    v8f;

__device__ __forceinline__ v8f vzero8() {
  v8f z;
#pragma unroll
  for (int i = 0; i < 8; ++i) z[i] = 0.0f;
  return z;
}

__device__ __forceinline__ v8f wmma_f16(v16h a, v16h b, v8f c) {
  // D = A(16x32 f16) * B(32x16 f16) + C(16x16 f32)
  return __builtin_amdgcn_wmma_f32_16x16x32_f16(
      /*neg_a=*/false, a, /*neg_b=*/false, b,
      /*c_mod=*/(short)0, c, /*reuse_a=*/false, /*reuse_b=*/false);
}

// A fragment: A row-major [16][lda] f16 in LDS, K-step k0 (multiple of 32).
// Lanes 0-15: M=lane, K = k0..k0+15
// Lanes 16-31: M=lane-16, K = k0+16..k0+31
__device__ __forceinline__ v16h frag_a(const _Float16* A, int lane, int lda, int k0) {
  int row = lane & 15, hi = lane >> 4;
  const _Float16* base = A + row * lda + k0 + hi * 16;
  union { float4 f4[2]; v16h v; } u;
  u.f4[0] = *(const float4*)(base);
  u.f4[1] = *(const float4*)(base + 8);
  return u.v;
}

// B fragment: BT row-major [16][ldb] f16 in LDS (row = output col N, col = K).
// Lanes 0-15: N=lane, K = k0..k0+15 ; Lanes 16-31: N=lane-16, K = k0+16..k0+31
__device__ __forceinline__ v16h frag_b(const _Float16* BT, int lane, int ldb, int k0) {
  int col = lane & 15, hi = lane >> 4;
  const _Float16* base = BT + col * ldb + k0 + hi * 16;
  union { float4 f4[2]; v16h v; } u;
  u.f4[0] = *(const float4*)(base);
  u.f4[1] = *(const float4*)(base + 8);
  return u.v;
}

// ---------- CDNA5 async global->LDS copy (ASYNCcnt path, inline asm) ----------
// LDS flat aperture: addr[31:0] is the LDS byte offset.
__device__ __forceinline__ void async_ld_b128(void* lds_dst, const void* gsrc) {
  unsigned lds_off = (unsigned)(unsigned long long)lds_dst;
  asm volatile("global_load_async_to_lds_b128 %0, %1, off"
               : : "v"(lds_off), "v"(gsrc) : "memory");
}
__device__ __forceinline__ void async_wait_all() {
  asm volatile("s_wait_asynccnt 0x0" : : : "memory");
}
// Cooperative f16 tile copy (halves must be a multiple of 8; 16B aligned both sides)
__device__ __forceinline__ void copy_f16_async(_Float16* dst, const _Float16* src,
                                               int halves, int tid, int nthreads) {
  for (int i = tid * 8; i < halves; i += nthreads * 8)
    async_ld_b128(dst + i, src + i);
}

// float atomic-max via integer punning (works with init = -inf bits)
__device__ __forceinline__ void atomicMaxF(float* addr, float v) {
  if (v >= 0.0f) atomicMax((int*)addr, __float_as_int(v));
  else           atomicMin((unsigned int*)addr, __float_as_uint(v));
}

#define NODE_DIM 32
#define HID 64
#define EDGE_DIM 7
#define KATT 160   // 2*HID + EDGE_DIM padded to multiple of 32

// ---------------- Kernel: pack / transpose weights to f16 ----------------
__global__ void prep_weights_kernel(const float* enc_w1, const float* enc_w2,
                                    const float* att_w1, const float* mrg_w,
                                    const float* out_w1,
                                    _Float16* enc_w1T, _Float16* enc_w2T,
                                    _Float16* att_w1T, _Float16* mrg_wT,
                                    _Float16* out_w1T) {
  int tid = blockIdx.x * blockDim.x + threadIdx.x;
  int nt  = gridDim.x * blockDim.x;
  for (int i = tid; i < 64 * 32; i += nt) {
    int n = i / 32, c = i % 32;
    enc_w1T[i] = (_Float16)enc_w1[c * 64 + n];
  }
  for (int i = tid; i < 64 * 64; i += nt) {
    int n = i / 64, c = i % 64;
    enc_w2T[i] = (_Float16)enc_w2[c * 64 + n];
  }
  for (int i = tid; i < 2 * 64 * KATT; i += nt) {
    int k = i / (64 * KATT), rem = i % (64 * KATT);
    int n = rem / KATT, c = rem % KATT;
    att_w1T[i] = (_Float16)((c < 135) ? att_w1[k * 135 * 64 + c * 64 + n] : 0.0f);
  }
  for (int i = tid; i < 2 * 64 * 128; i += nt) {
    int k = i / (64 * 128), rem = i % (64 * 128);
    int n = rem / 128, c = rem % 128;
    mrg_wT[i] = (_Float16)mrg_w[k * 128 * 64 + c * 64 + n];
  }
  for (int i = tid; i < 64 * 128; i += nt) {
    int n = i / 128, c = i % 128;
    out_w1T[i] = (_Float16)out_w1[c * 64 + n];
  }
}

// ---------------- Kernel: fused 2-layer node encoder (WMMA) ----------------
// h = relu(relu(x @ W1 + b1) @ W2 + b2), per block: 64 nodes (4 waves x 16)
__global__ void __launch_bounds__(128)
encoder_kernel(const float* __restrict__ x, const _Float16* __restrict__ w1T,
               const float* __restrict__ b1, const _Float16* __restrict__ w2T,
               const float* __restrict__ b2, float* __restrict__ h,
               _Float16* __restrict__ hf16, int N) {
  __shared__ __align__(32) _Float16 sX[4][16][NODE_DIM];
  __shared__ __align__(32) _Float16 sW1[64][NODE_DIM];
  __shared__ __align__(32) _Float16 sW2[64][HID];
  __shared__ __align__(32) _Float16 sH1[4][16][HID];
  __shared__ float sB1[64], sB2[64];

  int tid = threadIdx.x, wave = tid >> 5, lane = tid & 31;
  // weight tiles: pure f16 copies -> async global->LDS path
  copy_f16_async(&sW1[0][0], w1T, 64 * NODE_DIM, tid, 128);
  copy_f16_async(&sW2[0][0], w2T, 64 * HID, tid, 128);
  for (int i = tid; i < 64; i += blockDim.x) { sB1[i] = b1[i]; sB2[i] = b2[i]; }

  int rowbase = blockIdx.x * 64 + wave * 16;
  {
    int r = lane & 15, half = lane >> 4;
    int gr = rowbase + r;
#pragma unroll
    for (int c = 0; c < 16; ++c) {
      float v = (gr < N) ? x[(size_t)gr * NODE_DIM + half * 16 + c] : 0.0f;
      sX[wave][r][half * 16 + c] = (_Float16)v;
    }
  }
  async_wait_all();
  __syncthreads();

  int lo = lane & 15, hi = lane >> 4;
  // layer 1: K=32 (one WMMA step), N=64 (4 tiles)
  v8f acc[4];
#pragma unroll
  for (int t = 0; t < 4; ++t) acc[t] = vzero8();
  {
    v16h a = frag_a(&sX[wave][0][0], lane, NODE_DIM, 0);
#pragma unroll
    for (int t = 0; t < 4; ++t) {
      v16h b = frag_b(&sW1[t * 16][0], lane, NODE_DIM, 0);
      acc[t] = wmma_f16(a, b, acc[t]);
    }
  }
#pragma unroll
  for (int t = 0; t < 4; ++t) {
#pragma unroll
    for (int r = 0; r < 8; ++r) {
      int row = hi * 8 + r, col = t * 16 + lo;
      float v = acc[t][r] + sB1[col];
      sH1[wave][row][col] = (_Float16)(v > 0.0f ? v : 0.0f);
    }
  }
  __syncthreads();

  // layer 2: K=64 (two WMMA steps)
#pragma unroll
  for (int t = 0; t < 4; ++t) acc[t] = vzero8();
#pragma unroll
  for (int k0 = 0; k0 < HID; k0 += 32) {
    v16h a = frag_a(&sH1[wave][0][0], lane, HID, k0);
#pragma unroll
    for (int t = 0; t < 4; ++t) {
      v16h b = frag_b(&sW2[t * 16][0], lane, HID, k0);
      acc[t] = wmma_f16(a, b, acc[t]);
    }
  }
#pragma unroll
  for (int t = 0; t < 4; ++t) {
#pragma unroll
    for (int r = 0; r < 8; ++r) {
      int row = hi * 8 + r, col = t * 16 + lo;
      int gr = rowbase + row;
      if (gr < N) {
        float v = acc[t][r] + sB2[col];
        v = v > 0.0f ? v : 0.0f;
        h[(size_t)gr * HID + col] = v;
        hf16[(size_t)gr * HID + col] = (_Float16)v;
      }
    }
  }
}

// ---------------- Kernel: per-hop init ----------------
__global__ void init_hop_kernel(float* segmax, float* segsum, float* ctx, int N) {
  int i = blockIdx.x * blockDim.x + threadIdx.x;
  if (i < N) {
    segmax[i] = -__builtin_inff();
    segsum[i] = 0.0f;
  }
  if (i < N * HID) ctx[i] = 0.0f;
}

// ---------------- Kernel: edge attention MLP (WMMA) + score + segment max ----------------
// att_in[E,135->160] @ W1[160,64] -> relu -> dot with w2[64] -> score[E]
// Per block: 64 edges (4 waves x 16 edges). h gathers go global->LDS async.
__global__ void __launch_bounds__(128)
edge_score_kernel(const _Float16* __restrict__ hf16, const int* __restrict__ edge_index,
                  const float* __restrict__ edge_attr,
                  const _Float16* __restrict__ w1T, const float* __restrict__ b1,
                  const float* __restrict__ w2, const float* __restrict__ b2,
                  float* __restrict__ score, float* __restrict__ segmax, int E) {
  __shared__ __align__(32) _Float16 sA[4][16][KATT];   // 20 KB
  __shared__ __align__(32) _Float16 sB[64][KATT];      // 20 KB
  __shared__ float sB1[64], sW2[64];
  __shared__ int sDst[4][16];

  int tid = threadIdx.x, wave = tid >> 5, lane = tid & 31;
  copy_f16_async(&sB[0][0], w1T, 64 * KATT, tid, 128);
  for (int i = tid; i < 64; i += blockDim.x) { sB1[i] = b1[i]; sW2[i] = w2[i]; }

  int ebase = blockIdx.x * 64 + wave * 16;
  {
    int e = lane & 15, part = lane >> 4;   // 2 lanes cooperate per edge
    int ge = ebase + e;
    bool ok = ge < E;
    int s = 0, d = 0;
    if (ok) { s = edge_index[ge]; d = edge_index[E + ge]; }
    if (part == 0) sDst[wave][e] = ok ? d : 0;

    _Float16* a = &sA[wave][e][0];
    if (ok) {
      // h_i = h[dst] -> cols [0,64); h_j = h[src] -> cols [64,128); 32 halves per lane
      const _Float16* pd = hf16 + (size_t)d * HID + part * 32;
      const _Float16* ps = hf16 + (size_t)s * HID + part * 32;
      _Float16* qd = a + part * 32;
      _Float16* qs = a + 64 + part * 32;
#pragma unroll
      for (int i = 0; i < 4; ++i) async_ld_b128(qd + i * 8, pd + i * 8);
#pragma unroll
      for (int i = 0; i < 4; ++i) async_ld_b128(qs + i * 8, ps + i * 8);
    } else {
#pragma unroll
      for (int c = 0; c < 32; ++c) a[part * 32 + c] = (_Float16)0.0f;
#pragma unroll
      for (int c = 0; c < 32; ++c) a[64 + part * 32 + c] = (_Float16)0.0f;
    }
    // edge features + zero padding: cols [128,160)  (needs f32->f16 convert: VGPR path)
    if (part == 0) {
#pragma unroll
      for (int c = 0; c < 16; ++c) {
        float v = (ok && c < EDGE_DIM) ? edge_attr[(size_t)ge * EDGE_DIM + c] : 0.0f;
        a[128 + c] = (_Float16)v;
      }
    } else {
#pragma unroll
      for (int c = 0; c < 16; ++c) a[144 + c] = (_Float16)0.0f;
    }
  }
  async_wait_all();
  __syncthreads();

  v8f acc[4];
#pragma unroll
  for (int t = 0; t < 4; ++t) acc[t] = vzero8();
#pragma unroll
  for (int k0 = 0; k0 < KATT; k0 += 32) {
    v16h a = frag_a(&sA[wave][0][0], lane, KATT, k0);
#pragma unroll
    for (int t = 0; t < 4; ++t) {
      v16h b = frag_b(&sB[t * 16][0], lane, KATT, k0);
      acc[t] = wmma_f16(a, b, acc[t]);
    }
  }

  // score[m] = sum_n relu(hid[m,n] + b1[n]) * w2[n] ; reduce across 16-lane halves
  int lo = lane & 15, hi = lane >> 4;
  float sc[8];
#pragma unroll
  for (int r = 0; r < 8; ++r) sc[r] = 0.0f;
#pragma unroll
  for (int t = 0; t < 4; ++t) {
    int col = t * 16 + lo;
    float w = sW2[col], bb = sB1[col];
#pragma unroll
    for (int r = 0; r < 8; ++r) {
      float v = acc[t][r] + bb;
      v = v > 0.0f ? v : 0.0f;
      sc[r] += v * w;
    }
  }
#pragma unroll
  for (int off = 8; off > 0; off >>= 1) {
#pragma unroll
    for (int r = 0; r < 8; ++r) sc[r] += __shfl_xor(sc[r], off, 32);
  }
  if (lo == 0) {
    float bb2 = b2[0];
#pragma unroll
    for (int r = 0; r < 8; ++r) {
      int row = hi * 8 + r;
      int ge = ebase + row;
      if (ge < E) {
        float sv = sc[r] + bb2;
        score[ge] = sv;
        atomicMaxF(&segmax[sDst[wave][row]], sv);
      }
    }
  }
}

// ---------------- Kernel: ex = exp(score - max[dst]); segsum += ex ----------------
__global__ void softmax_ex_kernel(const int* __restrict__ edge_index,
                                  float* __restrict__ score, const float* __restrict__ segmax,
                                  float* __restrict__ segsum, int E) {
  int e = blockIdx.x * blockDim.x + threadIdx.x;
  if (e < E) {
    int d = edge_index[E + e];
    float ex = __expf(score[e] - segmax[d]);
    score[e] = ex;                         // reuse buffer for ex
    atomicAdd(&segsum[d], ex);
  }
}

// ---------------- Kernel: ctx[dst] += alpha * h[src] ----------------
__global__ void scatter_kernel(const int* __restrict__ edge_index,
                               const float* __restrict__ ex, const float* __restrict__ segsum,
                               const float* __restrict__ h, float* __restrict__ ctx, int E) {
  int t = blockIdx.x * blockDim.x + threadIdx.x;
  int e = t >> 3, part = t & 7;            // 8 threads per edge, 8 dims each
  if (e < E) {
    int s = edge_index[e], d = edge_index[E + e];
    float alpha = ex[e] / (segsum[d] + 1e-16f);
    const float* hj = h + (size_t)s * HID + part * 8;
    float* c = ctx + (size_t)d * HID + part * 8;
#pragma unroll
    for (int i = 0; i < 8; ++i) atomicAdd(&c[i], alpha * hj[i]);
  }
}

// ---------------- Kernel: merge  h' = relu([h|ctx] @ W + b)  (WMMA, K=128) ----------------
__global__ void __launch_bounds__(128)
merge_kernel(const float* __restrict__ h, const float* __restrict__ ctx,
             const _Float16* __restrict__ wT, const float* __restrict__ b,
             float* __restrict__ h_out, _Float16* __restrict__ hf16_out, int N) {
  __shared__ __align__(32) _Float16 sA[4][16][128];    // 16 KB
  __shared__ __align__(32) _Float16 sB[64][128];       // 16 KB
  __shared__ float sBias[64];

  int tid = threadIdx.x, wave = tid >> 5, lane = tid & 31;
  copy_f16_async(&sB[0][0], wT, 64 * 128, tid, 128);
  for (int i = tid; i < 64; i += blockDim.x) sBias[i] = b[i];

  int rowbase = blockIdx.x * 64 + wave * 16;
  {
    int r = lane & 15, part = lane >> 4;   // part0 -> h cols[0,64), part1 -> ctx cols[64,128)
    int gr = rowbase + r;
    const float* src = part ? (ctx + (size_t)(gr < N ? gr : 0) * HID)
                            : (h   + (size_t)(gr < N ? gr : 0) * HID);
#pragma unroll
    for (int c = 0; c < 64; ++c) {
      float v = (gr < N) ? src[c] : 0.0f;
      sA[wave][r][part * 64 + c] = (_Float16)v;
    }
  }
  async_wait_all();
  __syncthreads();

  v8f acc[4];
#pragma unroll
  for (int t = 0; t < 4; ++t) acc[t] = vzero8();
#pragma unroll
  for (int k0 = 0; k0 < 128; k0 += 32) {
    v16h a = frag_a(&sA[wave][0][0], lane, 128, k0);
#pragma unroll
    for (int t = 0; t < 4; ++t) {
      v16h bf = frag_b(&sB[t * 16][0], lane, 128, k0);
      acc[t] = wmma_f16(a, bf, acc[t]);
    }
  }
  int lo = lane & 15, hi = lane >> 4;
#pragma unroll
  for (int t = 0; t < 4; ++t) {
#pragma unroll
    for (int r = 0; r < 8; ++r) {
      int row = hi * 8 + r, col = t * 16 + lo;
      int gr = rowbase + row;
      if (gr < N) {
        float v = acc[t][r] + sBias[col];
        v = v > 0.0f ? v : 0.0f;
        h_out[(size_t)gr * HID + col] = v;
        hf16_out[(size_t)gr * HID + col] = (_Float16)v;
      }
    }
  }
}

// ---------------- Kernel: output head (WMMA K=128, then 64->2 dot) ----------------
__global__ void __launch_bounds__(128)
final_kernel(const float* __restrict__ h_local, const float* __restrict__ h,
             const _Float16* __restrict__ w1T, const float* __restrict__ b1,
             const float* __restrict__ w2, const float* __restrict__ b2,
             float* __restrict__ out, int N) {
  __shared__ __align__(32) _Float16 sA[4][16][128];
  __shared__ __align__(32) _Float16 sB[64][128];
  __shared__ float sB1[64], sW2[128];

  int tid = threadIdx.x, wave = tid >> 5, lane = tid & 31;
  copy_f16_async(&sB[0][0], w1T, 64 * 128, tid, 128);
  for (int i = tid; i < 64; i += blockDim.x) sB1[i] = b1[i];
  for (int i = tid; i < 128; i += blockDim.x) sW2[i] = w2[i];

  int rowbase = blockIdx.x * 64 + wave * 16;
  {
    int r = lane & 15, part = lane >> 4;   // part0 -> h_local, part1 -> h
    int gr = rowbase + r;
    const float* src = part ? (h       + (size_t)(gr < N ? gr : 0) * HID)
                            : (h_local + (size_t)(gr < N ? gr : 0) * HID);
#pragma unroll
    for (int c = 0; c < 64; ++c) {
      float v = (gr < N) ? src[c] : 0.0f;
      sA[wave][r][part * 64 + c] = (_Float16)v;
    }
  }
  async_wait_all();
  __syncthreads();

  v8f acc[4];
#pragma unroll
  for (int t = 0; t < 4; ++t) acc[t] = vzero8();
#pragma unroll
  for (int k0 = 0; k0 < 128; k0 += 32) {
    v16h a = frag_a(&sA[wave][0][0], lane, 128, k0);
#pragma unroll
    for (int t = 0; t < 4; ++t) {
      v16h bf = frag_b(&sB[t * 16][0], lane, 128, k0);
      acc[t] = wmma_f16(a, bf, acc[t]);
    }
  }
  // u = relu(acc + b1) ; out[:,j] = u @ w2[:,j] + b2[j]
  int lo = lane & 15, hi = lane >> 4;
  float s0[8], s1[8];
#pragma unroll
  for (int r = 0; r < 8; ++r) { s0[r] = 0.0f; s1[r] = 0.0f; }
#pragma unroll
  for (int t = 0; t < 4; ++t) {
    int col = t * 16 + lo;
    float w0 = sW2[col * 2 + 0], w1 = sW2[col * 2 + 1], bb = sB1[col];
#pragma unroll
    for (int r = 0; r < 8; ++r) {
      float v = acc[t][r] + bb;
      v = v > 0.0f ? v : 0.0f;
      s0[r] += v * w0;
      s1[r] += v * w1;
    }
  }
#pragma unroll
  for (int off = 8; off > 0; off >>= 1) {
#pragma unroll
    for (int r = 0; r < 8; ++r) {
      s0[r] += __shfl_xor(s0[r], off, 32);
      s1[r] += __shfl_xor(s1[r], off, 32);
    }
  }
  if (lo == 0) {
    float bb0 = b2[0], bb1 = b2[1];
#pragma unroll
    for (int r = 0; r < 8; ++r) {
      int gr = rowbase + hi * 8 + r;
      if (gr < N) {
        out[(size_t)gr * 2 + 0] = s0[r] + bb0;
        out[(size_t)gr * 2 + 1] = s1[r] + bb1;
      }
    }
  }
}

// ---------------- Host launch ----------------
extern "C" void kernel_launch(void* const* d_in, const int* in_sizes, int n_in,
                              void* d_out, int out_size, void* d_ws, size_t ws_size,
                              hipStream_t stream) {
  const float* x          = (const float*)d_in[0];
  const int*   edge_index = (const int*)d_in[1];
  const float* edge_attr  = (const float*)d_in[2];
  const float* enc_w1     = (const float*)d_in[3];
  const float* enc_b1     = (const float*)d_in[4];
  const float* enc_w2     = (const float*)d_in[5];
  const float* enc_b2     = (const float*)d_in[6];
  const float* att_w1     = (const float*)d_in[7];
  const float* att_b1     = (const float*)d_in[8];
  const float* att_w2     = (const float*)d_in[9];
  const float* att_b2     = (const float*)d_in[10];
  const float* mrg_w      = (const float*)d_in[11];
  const float* mrg_b      = (const float*)d_in[12];
  const float* out_w1     = (const float*)d_in[13];
  const float* out_b1     = (const float*)d_in[14];
  const float* out_w2     = (const float*)d_in[15];
  const float* out_b2     = (const float*)d_in[16];

  const int N = in_sizes[0] / NODE_DIM;
  const int E = in_sizes[1] / 2;

  // workspace carve-up
  char* base = (char*)d_ws;
  size_t off = 0;
  auto carve = [&](size_t bytes) {
    size_t o = off;
    off = (off + bytes + 255) & ~(size_t)255;
    return (void*)(base + o);
  };
  float*    h_enc   = (float*)carve((size_t)N * HID * 4);
  float*    hA      = (float*)carve((size_t)N * HID * 4);
  float*    hB      = (float*)carve((size_t)N * HID * 4);
  float*    ctx     = (float*)carve((size_t)N * HID * 4);
  _Float16* hf16    = (_Float16*)carve((size_t)N * HID * 2);
  float*    score   = (float*)carve((size_t)E * 4);
  float*    segmax  = (float*)carve((size_t)N * 4);
  float*    segsum  = (float*)carve((size_t)N * 4);
  _Float16* enc_w1T = (_Float16*)carve(64 * NODE_DIM * 2);
  _Float16* enc_w2T = (_Float16*)carve(64 * HID * 2);
  _Float16* att_w1T = (_Float16*)carve(2 * 64 * KATT * 2);
  _Float16* mrg_wT  = (_Float16*)carve(2 * 64 * 128 * 2);
  _Float16* out_w1T = (_Float16*)carve(64 * 128 * 2);
  (void)ws_size; (void)n_in; (void)out_size;

  // 1) pack weights
  prep_weights_kernel<<<64, 256, 0, stream>>>(enc_w1, enc_w2, att_w1, mrg_w, out_w1,
                                              enc_w1T, enc_w2T, att_w1T, mrg_wT, out_w1T);

  // 2) node encoder
  int node_blocks = (N + 63) / 64;
  encoder_kernel<<<node_blocks, 128, 0, stream>>>(x, enc_w1T, enc_b1, enc_w2T, enc_b2,
                                                  h_enc, hf16, N);

  // 3) hops
  float* h_prev = h_enc;
  float* bufs[2] = {hA, hB};
  int edge_blocks  = (E + 63) / 64;
  int init_blocks  = (N * HID + 255) / 256;
  int ex_blocks    = (E + 255) / 256;
  int scat_blocks  = (E * 8 + 255) / 256;
  for (int k = 0; k < 2; ++k) {
    init_hop_kernel<<<init_blocks, 256, 0, stream>>>(segmax, segsum, ctx, N);
    edge_score_kernel<<<edge_blocks, 128, 0, stream>>>(
        hf16, edge_index, edge_attr,
        att_w1T + (size_t)k * 64 * KATT, att_b1 + k * 64,
        att_w2 + k * 64, att_b2 + k, score, segmax, E);
    softmax_ex_kernel<<<ex_blocks, 256, 0, stream>>>(edge_index, score, segmax, segsum, E);
    scatter_kernel<<<scat_blocks, 256, 0, stream>>>(edge_index, score, segsum, h_prev, ctx, E);
    merge_kernel<<<node_blocks, 128, 0, stream>>>(h_prev, ctx,
                                                  mrg_wT + (size_t)k * 64 * 128, mrg_b + k * 64,
                                                  bufs[k], hf16, N);
    h_prev = bufs[k];
  }

  // 4) output head
  final_kernel<<<node_blocks, 128, 0, stream>>>(h_enc, h_prev, out_w1T, out_b1,
                                                out_w2, out_b2, (float*)d_out, N);
}
